// _RefInternLM3Attention_53472342835689
// MI455X (gfx1250) — compile-verified, matrix operand order checked by reading, not run on hardware
//
#include <hip/hip_runtime.h>

// ---------------------------------------------------------------------------
// InternLM3-style attention block for MI455X (gfx1250, wave32, WMMA).
// All matmuls run through v_wmma_f32_16x16x32_bf16 (fp32 accumulate).
// Working set (~130MB bf16) fits in the 192MB global L2, so GEMMs use
// per-lane 16B global loads in native WMMA fragment layout and rely on L2
// for reuse instead of LDS staging. Fragment loads are interleaved ahead of
// their consuming WMMAs (named scalars + #pragma unroll: no register-array
// movrels, no rolled loops). Flash attention streams 64-key blocks off the
// diagonal and 32-key masked blocks on the diagonal.
// ---------------------------------------------------------------------------

#define HIDDEN   4096
#define S_LEN    2048
#define NH       32
#define NKV      8
#define HD       128

typedef __attribute__((ext_vector_type(16))) __bf16 v16bf;
typedef __attribute__((ext_vector_type(8)))  float  v8f;

__device__ __forceinline__ unsigned short f2bf(float f) {
  unsigned int u = __float_as_uint(f);
  u += 0x7FFFu + ((u >> 16) & 1u);          // round-to-nearest-even
  return (unsigned short)(u >> 16);
}
__device__ __forceinline__ float bf2f(unsigned short h) {
  return __uint_as_float(((unsigned int)h) << 16);
}

union FragBF { uint4 u[2]; v16bf v; };

// 16x32 bf16 fragment, ISA layout: lanes 0-15 / 16-31 split K into
// group0 = half*8 .. half*8+7, group1 = 16+half*8 .. 16+half*8+7.
// rowptr = &row[kbase], must be 16B aligned at the group offsets.
__device__ __forceinline__ v16bf load_frag(const unsigned short* rowptr, int half) {
  FragBF f;
  f.u[0] = *(const uint4*)(rowptr + half * 8);
  f.u[1] = *(const uint4*)(rowptr + 16 + half * 8);
  return f.v;
}

__device__ __forceinline__ v8f wmma_bf16(v16bf a, v16bf b, v8f c) {
  return __builtin_amdgcn_wmma_f32_16x16x32_bf16(
      /*neg_a=*/false, a, /*neg_b=*/false, b,
      /*c_mod=*/(short)0, c, /*reuse_a=*/false, /*reuse_b=*/false);
}

__device__ __forceinline__ v8f vzero8() {
  v8f z = {0.f, 0.f, 0.f, 0.f, 0.f, 0.f, 0.f, 0.f};
  return z;
}

// ---------------------------------------------------------------------------
// fp32 -> bf16 conversion (bandwidth-trivial, done once per buffer)
// ---------------------------------------------------------------------------
__global__ void cvt_f32_bf16(const float* __restrict__ in,
                             unsigned short* __restrict__ out, long n) {
  long i = (long)blockIdx.x * blockDim.x + threadIdx.x;
  long stride = (long)gridDim.x * blockDim.x;
  for (; i < n; i += stride) out[i] = f2bf(in[i]);
}

// ---------------------------------------------------------------------------
// C[M,N] = X[M,K] * W[N,K]^T  (both row-major, K contiguous -> 16B frag loads)
// Block = 128 threads (4 waves), block tile 64x128; each wave 32(M)x64(N):
// 8 WMMAs per K=32 step, loads interleaved ahead of consumption.
// ---------------------------------------------------------------------------
template <bool OUT_F32>
__global__ __launch_bounds__(128) void gemm_bf16(
    const unsigned short* __restrict__ X,
    const unsigned short* __restrict__ W,
    void* __restrict__ Cout, int M, int N, int K) {
  const int lane = threadIdx.x & 31;
  const int wave = threadIdx.x >> 5;
  const int half = lane >> 4;
  const int lm   = lane & 15;

  const int m0 = blockIdx.x * 64 + (wave & 1) * 32;
  const int n0 = blockIdx.y * 128 + (wave >> 1) * 64;

  v8f acc0 = vzero8(), acc1 = vzero8(), acc2 = vzero8(), acc3 = vzero8();
  v8f acc4 = vzero8(), acc5 = vzero8(), acc6 = vzero8(), acc7 = vzero8();

  const unsigned short* xr0 = X + (size_t)(m0 + lm) * K;
  const unsigned short* xr1 = X + (size_t)(m0 + 16 + lm) * K;
  const unsigned short* wr0 = W + (size_t)(n0 + lm) * K;
  const unsigned short* wr1 = W + (size_t)(n0 + 16 + lm) * K;
  const unsigned short* wr2 = W + (size_t)(n0 + 32 + lm) * K;
  const unsigned short* wr3 = W + (size_t)(n0 + 48 + lm) * K;

  for (int k = 0; k < K; k += 32) {
    // pull the next K-chunk of the streaming operands into the WGP cache
    if (k + 256 < K) {
      __builtin_prefetch(xr0 + k + 256, 0, 3);
      __builtin_prefetch(wr0 + k + 256, 0, 3);
      __builtin_prefetch(wr2 + k + 256, 0, 3);
    }
    // interleave: keep several b128 loads in flight ahead of each WMMA
    v16bf a0 = load_frag(xr0 + k, half);
    v16bf a1 = load_frag(xr1 + k, half);
    v16bf b0 = load_frag(wr0 + k, half);
    v16bf b1 = load_frag(wr1 + k, half);
    acc0 = wmma_bf16(a0, b0, acc0);
    v16bf b2 = load_frag(wr2 + k, half);
    acc4 = wmma_bf16(a1, b0, acc4);
    v16bf b3 = load_frag(wr3 + k, half);
    acc1 = wmma_bf16(a0, b1, acc1);
    acc5 = wmma_bf16(a1, b1, acc5);
    acc2 = wmma_bf16(a0, b2, acc2);
    acc6 = wmma_bf16(a1, b2, acc6);
    acc3 = wmma_bf16(a0, b3, acc3);
    acc7 = wmma_bf16(a1, b3, acc7);
  }

  // C layout: lane holds N = n0+nt*16+lm ; VGPR r holds M = r + 8*half.
  v8f accs[2][4] = {{acc0, acc1, acc2, acc3}, {acc4, acc5, acc6, acc7}};
#pragma unroll
  for (int mt = 0; mt < 2; ++mt)
#pragma unroll
    for (int nt = 0; nt < 4; ++nt)
#pragma unroll
      for (int r = 0; r < 8; ++r) {
        int m = m0 + mt * 16 + r + 8 * half;
        int n = n0 + nt * 16 + lm;
        float v = accs[mt][nt][r];
        if (OUT_F32) ((float*)Cout)[(size_t)m * N + n] = v;
        else ((unsigned short*)Cout)[(size_t)m * N + n] = f2bf(v);
      }
}

// ---------------------------------------------------------------------------
// In-place RoPE on a [S, heads*128] bf16 buffer. One thread per (s,h,d<64).
// out[d]    = (x[d]*cos - x[d+64]*sin) * outscale
// out[d+64] = (x[d+64]*cos + x[d]*sin) * outscale
// outscale folds the attention 1/sqrt(HD) into Q (1.0 for K).
// ---------------------------------------------------------------------------
__global__ void rope_kernel(unsigned short* __restrict__ buf,
                            const int* __restrict__ pos,
                            int heads, int S, float outscale) {
  int total = S * heads * 64;
  for (int i = blockIdx.x * blockDim.x + threadIdx.x; i < total;
       i += gridDim.x * blockDim.x) {
    int d = i & 63;
    int t = i >> 6;
    int h = t % heads;
    int s = t / heads;
    // inv_freq = 10000^(-2d/128) = exp(-d * ln(10000)/64)
    float inv = __expf(-(float)d * 0.14391156514261f);
    float ang = (float)pos[s] * inv;
    float c  = __cosf(ang);
    float sn = __sinf(ang);
    size_t base = (size_t)s * (heads * HD) + (size_t)h * HD + d;
    float x0 = bf2f(buf[base]);
    float x1 = bf2f(buf[base + 64]);
    buf[base]      = f2bf((x0 * c - x1 * sn) * outscale);
    buf[base + 64] = f2bf((x1 * c + x0 * sn) * outscale);
  }
}

// ---------------------------------------------------------------------------
// Vt[c, r] = V[r, c] : makes P*V B-fragments K(=key)-contiguous in memory.
// ---------------------------------------------------------------------------
__global__ void transpose_bf16(const unsigned short* __restrict__ in,
                               unsigned short* __restrict__ out,
                               int R, int C) {
  long total = (long)R * C;
  for (long i = (long)blockIdx.x * blockDim.x + threadIdx.x; i < total;
       i += (long)gridDim.x * blockDim.x) {
    int r = (int)(i / C);
    int c = (int)(i % C);
    out[(size_t)c * R + r] = in[i];
  }
}

// ---------------------------------------------------------------------------
// 64-key unmasked flash block: 16 score WMMAs + ONE softmax reduction tree +
// one oacc rescale + 16 PV WMMAs. ptile stride is 64 columns.
// ---------------------------------------------------------------------------
__device__ __forceinline__ void fa_block64(
    int n0, int lm, int half, int kvh, int S,
    const unsigned short* __restrict__ Kb,
    const unsigned short* __restrict__ Vt,
    const v16bf (&qf)[4], v8f (&oacc)[8],
    float (&rmax)[8], float (&rsum)[8],
    unsigned short* ptile) {
  // ---- scores: four 16x16 N-subtiles, K-dim 128 ----
  v8f s0 = vzero8(), s1 = vzero8(), s2 = vzero8(), s3 = vzero8();
  const unsigned short* kr0 = Kb + (size_t)(n0 + lm)      * (NKV * HD) + (size_t)kvh * HD;
  const unsigned short* kr1 = Kb + (size_t)(n0 + 16 + lm) * (NKV * HD) + (size_t)kvh * HD;
  const unsigned short* kr2 = Kb + (size_t)(n0 + 32 + lm) * (NKV * HD) + (size_t)kvh * HD;
  const unsigned short* kr3 = Kb + (size_t)(n0 + 48 + lm) * (NKV * HD) + (size_t)kvh * HD;
#pragma unroll
  for (int f = 0; f < 4; ++f) {
    v16bf b0 = load_frag(kr0 + f * 32, half);
    v16bf b1 = load_frag(kr1 + f * 32, half);
    s0 = wmma_bf16(qf[f], b0, s0);
    v16bf b2 = load_frag(kr2 + f * 32, half);
    s1 = wmma_bf16(qf[f], b1, s1);
    v16bf b3 = load_frag(kr3 + f * 32, half);
    s2 = wmma_bf16(qf[f], b2, s2);
    s3 = wmma_bf16(qf[f], b3, s3);
  }

  __syncthreads();   // previous block finished reading ptile
#pragma unroll
  for (int r = 0; r < 8; ++r) {
    float v0 = s0[r], v1 = s1[r], v2 = s2[r], v3 = s3[r];
    float bm = fmaxf(fmaxf(v0, v1), fmaxf(v2, v3));
#pragma unroll
    for (int off = 1; off < 16; off <<= 1)
      bm = fmaxf(bm, __shfl_xor(bm, off, 32));
    float nm = fmaxf(rmax[r], bm);
    float sc = __expf(rmax[r] - nm);
    rmax[r] = nm;
    float e0 = __expf(v0 - nm);
    float e1 = __expf(v1 - nm);
    float e2 = __expf(v2 - nm);
    float e3 = __expf(v3 - nm);
    float bs = (e0 + e1) + (e2 + e3);
#pragma unroll
    for (int off = 1; off < 16; off <<= 1)
      bs += __shfl_xor(bs, off, 32);
    rsum[r] = rsum[r] * sc + bs;
#pragma unroll
    for (int t = 0; t < 8; ++t) oacc[t][r] *= sc;
    int row = r + 8 * half;
    ptile[row * 64 + lm]      = f2bf(e0);
    ptile[row * 64 + 16 + lm] = f2bf(e1);
    ptile[row * 64 + 32 + lm] = f2bf(e2);
    ptile[row * 64 + 48 + lm] = f2bf(e3);
  }
  __syncthreads();
  FragBF pa0, pa1;
  pa0.u[0] = *(const uint4*)(ptile + lm * 64 + half * 8);
  pa0.u[1] = *(const uint4*)(ptile + lm * 64 + 16 + half * 8);
  pa1.u[0] = *(const uint4*)(ptile + lm * 64 + 32 + half * 8);
  pa1.u[1] = *(const uint4*)(ptile + lm * 64 + 48 + half * 8);

  // ---- O += P x V over 64 keys ----
#pragma unroll
  for (int t = 0; t < 8; t += 2) {
    const unsigned short* vr0 =
        Vt + (size_t)(kvh * HD + t * 16 + lm) * S + n0;
    const unsigned short* vr1 =
        Vt + (size_t)(kvh * HD + (t + 1) * 16 + lm) * S + n0;
    v16bf vb00 = load_frag(vr0, half);
    v16bf vb01 = load_frag(vr0 + 32, half);
    oacc[t] = wmma_bf16(pa0.v, vb00, oacc[t]);
    v16bf vb10 = load_frag(vr1, half);
    oacc[t] = wmma_bf16(pa1.v, vb01, oacc[t]);
    v16bf vb11 = load_frag(vr1 + 32, half);
    oacc[t + 1] = wmma_bf16(pa0.v, vb10, oacc[t + 1]);
    oacc[t + 1] = wmma_bf16(pa1.v, vb11, oacc[t + 1]);
  }
}

// ---------------------------------------------------------------------------
// 32-key flash block (diagonal path). MASKED applies the causal cndmask.
// ---------------------------------------------------------------------------
template <bool MASKED>
__device__ __forceinline__ void fa_block32(
    int n0, int m0, int lm, int half, int kvh, int S,
    const unsigned short* __restrict__ Kb,
    const unsigned short* __restrict__ Vt,
    const v16bf (&qf)[4], v8f (&oacc)[8],
    float (&rmax)[8], float (&rsum)[8],
    unsigned short* ptile) {
  v8f s0 = vzero8(), s1 = vzero8();
  const unsigned short* kr0 = Kb + (size_t)(n0 + lm)      * (NKV * HD) + (size_t)kvh * HD;
  const unsigned short* kr1 = Kb + (size_t)(n0 + 16 + lm) * (NKV * HD) + (size_t)kvh * HD;
#pragma unroll
  for (int f = 0; f < 4; ++f) {
    v16bf b0 = load_frag(kr0 + f * 32, half);
    v16bf b1 = load_frag(kr1 + f * 32, half);
    s0 = wmma_bf16(qf[f], b0, s0);
    s1 = wmma_bf16(qf[f], b1, s1);
  }

  __syncthreads();
#pragma unroll
  for (int r = 0; r < 8; ++r) {
    float v0 = s0[r];
    float v1 = s1[r];
    if (MASKED) {
      int m   = m0 + r + 8 * half;
      int kp0 = n0 + lm;
      int kp1 = n0 + 16 + lm;
      v0 = (kp0 <= m) ? v0 : -1e30f;
      v1 = (kp1 <= m) ? v1 : -1e30f;
    }
    float bm = fmaxf(v0, v1);
#pragma unroll
    for (int off = 1; off < 16; off <<= 1)
      bm = fmaxf(bm, __shfl_xor(bm, off, 32));
    float nm = fmaxf(rmax[r], bm);
    float sc = __expf(rmax[r] - nm);
    rmax[r] = nm;
    float e0 = __expf(v0 - nm);
    float e1 = __expf(v1 - nm);
    float bs = e0 + e1;
#pragma unroll
    for (int off = 1; off < 16; off <<= 1)
      bs += __shfl_xor(bs, off, 32);
    rsum[r] = rsum[r] * sc + bs;
#pragma unroll
    for (int t = 0; t < 8; ++t) oacc[t][r] *= sc;
    int row = r + 8 * half;
    ptile[row * 64 + lm]      = f2bf(e0);
    ptile[row * 64 + 16 + lm] = f2bf(e1);
  }
  __syncthreads();
  FragBF pa;
  pa.u[0] = *(const uint4*)(ptile + lm * 64 + half * 8);
  pa.u[1] = *(const uint4*)(ptile + lm * 64 + 16 + half * 8);

#pragma unroll
  for (int t = 0; t < 8; t += 2) {
    const unsigned short* vr0 =
        Vt + (size_t)(kvh * HD + t * 16 + lm) * S + n0;
    const unsigned short* vr1 =
        Vt + (size_t)(kvh * HD + (t + 1) * 16 + lm) * S + n0;
    v16bf vb0 = load_frag(vr0, half);
    v16bf vb1 = load_frag(vr1, half);
    oacc[t]     = wmma_bf16(pa.v, vb0, oacc[t]);
    oacc[t + 1] = wmma_bf16(pa.v, vb1, oacc[t + 1]);
  }
}

// ---------------------------------------------------------------------------
// Causal flash attention, one wave (32 threads) per 16-row Q tile per head.
// floor(m0/64) unmasked 64-key blocks, then <=1 unmasked 32-key block, then
// 1-2 masked 32-key diagonal blocks.
// ---------------------------------------------------------------------------
__global__ __launch_bounds__(32) void flash_attn(
    const unsigned short* __restrict__ Q,   // [S, NH*HD] bf16 (RoPE+scale)
    const unsigned short* __restrict__ Kb,  // [S, NKV*HD] bf16 (RoPE)
    const unsigned short* __restrict__ Vt,  // [NKV*HD, S] bf16 (transposed)
    unsigned short* __restrict__ O,         // [S, NH*HD] bf16
    int S) {
  __shared__ __attribute__((aligned(16))) unsigned short ptile[16 * 64];

  const int lane = threadIdx.x;
  const int half = lane >> 4;
  const int lm   = lane & 15;
  const int ntiles = S / 16;
  const int qt = blockIdx.x % ntiles;
  const int h  = blockIdx.x / ntiles;
  const int m0 = qt * 16;
  const int kvh = h >> 2;                   // GQA: 4 query heads per KV head

  // Q fragments: A-matrix 16x128 = four 16x32 frags
  v16bf qf[4];
  {
    const unsigned short* qrow = Q + (size_t)(m0 + lm) * (NH * HD) + (size_t)h * HD;
#pragma unroll
    for (int f = 0; f < 4; ++f) qf[f] = load_frag(qrow + f * 32, half);
  }

  v8f oacc[8];
#pragma unroll
  for (int t = 0; t < 8; ++t) oacc[t] = vzero8();
  float rmax[8], rsum[8];
#pragma unroll
  for (int r = 0; r < 8; ++r) { rmax[r] = -1e30f; rsum[r] = 0.f; }

  int n0 = 0;
  for (; n0 + 63 <= m0; n0 += 64)
    fa_block64(n0, lm, half, kvh, S, Kb, Vt, qf, oacc, rmax, rsum, ptile);
  for (; n0 + 31 <= m0; n0 += 32)
    fa_block32<false>(n0, m0, lm, half, kvh, S, Kb, Vt, qf, oacc, rmax, rsum, ptile);
  for (; n0 < m0 + 16; n0 += 32)
    fa_block32<true>(n0, m0, lm, half, kvh, S, Kb, Vt, qf, oacc, rmax, rsum, ptile);

  // ---- normalize and store ----
#pragma unroll
  for (int t = 0; t < 8; ++t)
#pragma unroll
    for (int r = 0; r < 8; ++r) {
      int m = m0 + r + 8 * half;
      int n = h * HD + t * 16 + lm;
      O[(size_t)m * (NH * HD) + n] = f2bf(oacc[t][r] / rsum[r]);
    }
}

// ---------------------------------------------------------------------------
// Host-side orchestration
// ---------------------------------------------------------------------------
extern "C" void kernel_launch(void* const* d_in, const int* in_sizes, int n_in,
                              void* d_out, int out_size, void* d_ws, size_t ws_size,
                              hipStream_t stream) {
  const float* hs  = (const float*)d_in[0];   // [1, 2048, 4096]
  const float* qw  = (const float*)d_in[1];   // [4096, 4096]
  const float* kw  = (const float*)d_in[2];   // [1024, 4096]
  const float* vw  = (const float*)d_in[3];   // [1024, 4096]
  const float* ow  = (const float*)d_in[4];   // [4096, 4096]
  const int*   pid = (const int*)d_in[5];     // [1, 2048]
  float* out = (float*)d_out;

  // workspace layout (bytes, all 16B-aligned)
  char* ws = (char*)d_ws;
  unsigned short* Xb  = (unsigned short*)(ws);                         // 16 MB
  unsigned short* Wqb = (unsigned short*)(ws + (16UL << 20));          // 32 MB
  unsigned short* Wkb = (unsigned short*)(ws + (48UL << 20));          //  8 MB
  unsigned short* Wvb = (unsigned short*)(ws + (56UL << 20));          //  8 MB
  unsigned short* Wob = (unsigned short*)(ws + (64UL << 20));          // 32 MB
  unsigned short* Qb  = (unsigned short*)(ws + (96UL << 20));          // 16 MB
  unsigned short* Kb  = (unsigned short*)(ws + (112UL << 20));         //  4 MB
  unsigned short* Vb  = (unsigned short*)(ws + (116UL << 20));         //  4 MB
  unsigned short* Vtb = (unsigned short*)(ws + (120UL << 20));         //  4 MB
  unsigned short* Ab  = (unsigned short*)(ws + (124UL << 20));         // 16 MB

  // 1) fp32 -> bf16
  cvt_f32_bf16<<<2048, 256, 0, stream>>>(hs, Xb, (long)S_LEN * HIDDEN);
  cvt_f32_bf16<<<2048, 256, 0, stream>>>(qw, Wqb, (long)NH * HD * HIDDEN);
  cvt_f32_bf16<<<2048, 256, 0, stream>>>(kw, Wkb, (long)NKV * HD * HIDDEN);
  cvt_f32_bf16<<<2048, 256, 0, stream>>>(vw, Wvb, (long)NKV * HD * HIDDEN);
  cvt_f32_bf16<<<2048, 256, 0, stream>>>(ow, Wob, (long)HIDDEN * NH * HD);

  // 2) QKV projections (bf16 WMMA, fp32 accumulate, bf16 out)
  gemm_bf16<false><<<dim3(S_LEN / 64, (NH * HD) / 128), 128, 0, stream>>>(
      Xb, Wqb, Qb, S_LEN, NH * HD, HIDDEN);
  gemm_bf16<false><<<dim3(S_LEN / 64, (NKV * HD) / 128), 128, 0, stream>>>(
      Xb, Wkb, Kb, S_LEN, NKV * HD, HIDDEN);
  gemm_bf16<false><<<dim3(S_LEN / 64, (NKV * HD) / 128), 128, 0, stream>>>(
      Xb, Wvb, Vb, S_LEN, NKV * HD, HIDDEN);

  // 3) RoPE on Q (with fused 1/sqrt(HD)) and K (in place)
  rope_kernel<<<2048, 256, 0, stream>>>(Qb, pid, NH, S_LEN, 0.08838834764831845f);
  rope_kernel<<<512, 256, 0, stream>>>(Kb, pid, NKV, S_LEN, 1.0f);

  // 4) V^T so P*V fragments are key-contiguous
  transpose_bf16<<<2048, 256, 0, stream>>>(Vb, Vtb, S_LEN, NKV * HD);

  // 5) causal flash attention (one wave per 16-row tile per head)
  flash_attn<<<NH * (S_LEN / 16), 32, 0, stream>>>(Qb, Kb, Vtb, Ab, S_LEN);

  // 6) output projection -> fp32
  gemm_bf16<true><<<dim3(S_LEN / 64, HIDDEN / 128), 128, 0, stream>>>(
      Ab, Wob, out, S_LEN, HIDDEN, NH * HD);
}
